// AdditiveAttention_82927228551222
// MI455X (gfx1250) — matrix operand order repaired and checked by reading, not compiled, not run on hardware
//
#include <hip/hip_runtime.h>
#include <hip/hip_bf16.h>
#include <math.h>

typedef __attribute__((ext_vector_type(16))) __bf16        v16bf;
typedef __attribute__((ext_vector_type(8)))  float         v8f;
typedef __attribute__((ext_vector_type(4)))  unsigned int  v4u;

#define B_   32
#define TK_  4096
#define H_   1024
#define A_   1024
#define MT_  32    // key rows per block (2 WMMA M-subtiles)

__device__ __forceinline__ unsigned short f2bf(float f) {
  unsigned int u = __float_as_uint(f);
  u += 0x7FFFu + ((u >> 16) & 1u);   // round-to-nearest-even
  return (unsigned short)(u >> 16);
}

union FragU { v4u u[2]; v16bf v; };

// ---------- kernel 0: q = query @ Wq  -> qws[B, A] (67 MFLOP, trivial) ----------
__global__ __launch_bounds__(256) void k_qproj(const float* __restrict__ query,
                                               const float* __restrict__ Wq,
                                               float* __restrict__ qws) {
  int gid = blockIdx.x * 256 + threadIdx.x;          // B*A threads
  int b = gid >> 10, a = gid & (A_ - 1);
  const float* qr = query + b * H_;
  float acc = 0.f;
  for (int h = 0; h < H_; ++h) acc = fmaf(qr[h], Wq[h * A_ + a], acc);
  qws[gid] = acc;
}

// ---------- kernel 1: repack Wk f32[H,A] -> bf16 in WMMA B-fragment layout ----------
// B 32x16 bf16 as packed here (contiguous k, same as A): lane = col + 16*(K>=16 half); per-lane bf16 j => K = j + 16*half.
// Layout: Bst[tileA(64)][chunk c(32)][lane(32)][j(16)] -> 32B/lane, 1KB per (tile,c), coalesced.
__global__ __launch_bounds__(256) void k_bstage(const float* __restrict__ Wk,
                                                unsigned short* __restrict__ Bst) {
  int gid = blockIdx.x * 256 + threadIdx.x;          // H*A threads, coalesced read of Wk
  int K = gid >> 10, a = gid & (A_ - 1);
  int h = (K >> 4) & 1, j = K & 15, c = K >> 5;
  int lane = (a & 15) + 16 * h, tileA = a >> 4;
  Bst[(((tileA * 32 + c) * 32 + lane) << 4) + j] = f2bf(Wk[gid]);
}

// ---------- kernel 2: fused k = keys@Wk, scores = v . tanh(q + k), masked ----------
// One block = (batch b, 32 key rows). Per wave: 2 M-subtiles x 8 N-tiles x 32 K-chunks,
// each B fragment feeds 2 WMMAs (halves L2 traffic vs 16-row tiles).
__global__ __launch_bounds__(256) void k_scores(const float* __restrict__ keys,
                                                const int* __restrict__ mask,
                                                const float* __restrict__ qws,
                                                const unsigned short* __restrict__ Bst,
                                                const float* __restrict__ vvec,
                                                float* __restrict__ scores) {
  __shared__ unsigned short Ast[2 * 32 * 32 * 16];   // 64 KB: A fragments for 2 M-subtiles
  __shared__ float sc[MT_];
  int tid = threadIdx.x;
  int b = blockIdx.x >> 7;                           // 128 row-tiles per batch
  int t0 = (blockIdx.x & 127) << 5;

  // Stage 32x1024 keys tile (f32, coalesced float4) -> bf16 A-fragment order in LDS.
  // A 16x32 bf16 as packed here (contiguous k, same as B): lane = row + 16*h; per-lane j: K = j + 16*h.
  const float* kp = keys + ((size_t)b * TK_ + t0) * H_;
  for (int i = 0; i < 32; ++i) {
    int f4 = tid + i * 256;
    int row = f4 >> 8;                               // 0..31
    int col0 = (f4 & 255) * 4;
    float4 val = *(const float4*)(kp + row * H_ + col0);
    float e[4] = {val.x, val.y, val.z, val.w};
    int mrow = row >> 4, r16 = row & 15;
    #pragma unroll
    for (int q = 0; q < 4; ++q) {
      int col = col0 + q;
      int c = col >> 5, kk = col & 31;
      int h, j;
      h = kk >> 4; j = kk & 15;
      Ast[(((mrow * 32 + c) * 32) + r16 + 16 * h) * 16 + j] = f2bf(e[q]);
    }
  }
  if (tid < MT_) sc[tid] = 0.f;
  __syncthreads();

  int wave = tid >> 5, lane = tid & 31, ln = lane & 15, half = lane >> 4;
  v8f acc[2][8] = {};                                // 2 M x 8 N tiles of 16x16 f32 per wave
  const unsigned short* bbase = Bst + (size_t)(wave * 8) * 32 * 32 * 16;
  for (int c = 0; c < 32; ++c) {                     // K chunks of 32
    FragU af0, af1;
    const v4u* ap0 = (const v4u*)&Ast[((0 * 32 + c) * 32 + lane) * 16];
    const v4u* ap1 = (const v4u*)&Ast[((1 * 32 + c) * 32 + lane) * 16];
    af0.u[0] = ap0[0]; af0.u[1] = ap0[1];
    af1.u[0] = ap1[0]; af1.u[1] = ap1[1];
    #pragma unroll
    for (int t = 0; t < 8; ++t) {
      FragU bfg;
      const v4u* bp = (const v4u*)(bbase + (size_t)(((t * 32 + c) * 32 + lane) << 4));
      bfg.u[0] = bp[0]; bfg.u[1] = bp[1];
      acc[0][t] = __builtin_amdgcn_wmma_f32_16x16x32_bf16(false, af0.v, false, bfg.v,
                                                          (short)0, acc[0][t], false, false);
      acc[1][t] = __builtin_amdgcn_wmma_f32_16x16x32_bf16(false, af1.v, false, bfg.v,
                                                          (short)0, acc[1][t], false, false);
    }
  }

  // Epilogue: scores_m = sum_a v[a]*tanh(q[a] + k[m][a]); D: VGPR r <-> M=r+8*half, N=ln.
  float part[2][8];
  #pragma unroll
  for (int m = 0; m < 2; ++m)
    #pragma unroll
    for (int r = 0; r < 8; ++r) part[m][r] = 0.f;
  #pragma unroll
  for (int t = 0; t < 8; ++t) {
    int a0 = (wave * 8 + t) * 16 + ln;
    float qv = qws[b * A_ + a0];
    float vv = vvec[a0];
    #pragma unroll
    for (int m = 0; m < 2; ++m)
      #pragma unroll
      for (int r = 0; r < 8; ++r) part[m][r] += vv * tanhf(qv + acc[m][t][r]);
  }
  #pragma unroll
  for (int s = 1; s < 16; s <<= 1) {
    #pragma unroll
    for (int m = 0; m < 2; ++m)
      #pragma unroll
      for (int r = 0; r < 8; ++r) part[m][r] += __shfl_xor(part[m][r], s, 32);
  }
  if (ln == 0) {
    #pragma unroll
    for (int m = 0; m < 2; ++m)
      #pragma unroll
      for (int r = 0; r < 8; ++r) atomicAdd(&sc[m * 16 + 8 * half + r], part[m][r]);
  }
  __syncthreads();
  if (tid < MT_) {
    int t = t0 + tid;
    bool ok = mask[b * TK_ + t] != 0;
    scores[b * TK_ + t] = ok ? sc[tid] : -__builtin_inff();
  }
}

// ---------- kernel 3: masked softmax + ctx = attn^T . values (memory-bound) ----------
__global__ __launch_bounds__(256) void k_finalize(const float* __restrict__ values,
                                                  float* __restrict__ attn,   // raw scores on entry
                                                  float* __restrict__ ctx) {
  __shared__ float sa[TK_];                          // 16 KB
  __shared__ float red[8];
  int tid = threadIdx.x, b = blockIdx.x;
  int wave = tid >> 5, lane = tid & 31;

  float lmax = -__builtin_inff();
  for (int i = 0; i < 16; ++i) {
    int x = tid + i * 256;
    float s = attn[b * TK_ + x];
    sa[x] = s;
    lmax = fmaxf(lmax, s);
  }
  #pragma unroll
  for (int m = 1; m < 32; m <<= 1) lmax = fmaxf(lmax, __shfl_xor(lmax, m, 32));
  if (lane == 0) red[wave] = lmax;
  __syncthreads();
  float bmax = red[0];
  #pragma unroll
  for (int w = 1; w < 8; ++w) bmax = fmaxf(bmax, red[w]);
  __syncthreads();

  float lsum = 0.f;
  for (int i = 0; i < 16; ++i) {
    int x = tid + i * 256;
    float e = __expf(sa[x] - bmax);                  // -inf -> 0
    sa[x] = e;
    lsum += e;
  }
  #pragma unroll
  for (int m = 1; m < 32; m <<= 1) lsum += __shfl_xor(lsum, m, 32);
  if (lane == 0) red[wave] = lsum;
  __syncthreads();
  float tot = 0.f;
  #pragma unroll
  for (int w = 0; w < 8; ++w) tot += red[w];
  float inv = 1.f / tot;

  for (int i = 0; i < 16; ++i) {
    int x = tid + i * 256;
    float aw = sa[x] * inv;
    sa[x] = aw;
    attn[b * TK_ + x] = aw;
  }
  __syncthreads();

  float c0 = 0.f, c1 = 0.f, c2 = 0.f, c3 = 0.f;
  const float* vb = values + (size_t)b * TK_ * H_;
  for (int t = 0; t < TK_; ++t) {
    float w = sa[t];
    if (w > 0.f) {                                   // skip masked rows: halves HBM traffic
      const float* vr = vb + (size_t)t * H_;
      c0 = fmaf(w, vr[tid      ], c0);
      c1 = fmaf(w, vr[tid + 256], c1);
      c2 = fmaf(w, vr[tid + 512], c2);
      c3 = fmaf(w, vr[tid + 768], c3);
    }
  }
  ctx[b * H_ + tid      ] = c0;
  ctx[b * H_ + tid + 256] = c1;
  ctx[b * H_ + tid + 512] = c2;
  ctx[b * H_ + tid + 768] = c3;
}

extern "C" void kernel_launch(void* const* d_in, const int* in_sizes, int n_in,
                              void* d_out, int out_size, void* d_ws, size_t ws_size,
                              hipStream_t stream) {
  (void)in_sizes; (void)n_in; (void)out_size; (void)ws_size;
  const float* query  = (const float*)d_in[0];
  const float* keys   = (const float*)d_in[1];
  const float* values = (const float*)d_in[2];
  const int*   mask   = (const int*)  d_in[3];
  const float* Wq     = (const float*)d_in[4];
  const float* Wk     = (const float*)d_in[5];
  const float* vvec   = (const float*)d_in[6];

  float* ctx  = (float*)d_out;                       // [B,1,H]
  float* attn = (float*)d_out + B_ * H_;             // [B,1,TK] (scores written here first)

  float* qws = (float*)d_ws;                                             // 128 KB
  unsigned short* Bst = (unsigned short*)((char*)d_ws + B_ * A_ * 4);    // 2 MB bf16 Wk fragments

  k_qproj   <<<dim3(B_ * A_ / 256),    dim3(256), 0, stream>>>(query, Wq, qws);
  k_bstage  <<<dim3(H_ * A_ / 256),    dim3(256), 0, stream>>>(Wk, Bst);
  k_scores  <<<dim3(B_ * (TK_ / MT_)), dim3(256), 0, stream>>>(keys, mask, qws, Bst, vvec, attn);
  k_finalize<<<dim3(B_),               dim3(256), 0, stream>>>(values, attn, ctx);
}
